// GraphPropLayer_21105469293020
// MI455X (gfx1250) — matrix operand order repaired and checked by reading, not compiled, not run on hardware
//
#include <hip/hip_runtime.h>
#include <stdint.h>

typedef unsigned short ushort_t;
typedef __attribute__((ext_vector_type(16))) __bf16   v16bf;
typedef __attribute__((ext_vector_type(8)))  float    v8f;
typedef __attribute__((ext_vector_type(4)))  unsigned u32x4;
typedef int v4i __attribute__((vector_size(16)));

union Frag { v16bf v; u32x4 u[2]; };

#if __has_builtin(__builtin_amdgcn_global_load_async_to_lds_b128)
#define USE_ASYNC_LDS 1
// exact param types per hipcc diagnostic: AS1 (printed __device__) v4i*, AS3 v4i*
#define GPTRI(p) ((__attribute__((address_space(1))) v4i*)(p))
#define LPTRI(p) ((__attribute__((address_space(3))) v4i*)(p))
#endif

__device__ __forceinline__ void wait_async_zero() {
#if defined(USE_ASYNC_LDS)
#if __has_builtin(__builtin_amdgcn_s_wait_asynccnt)
  __builtin_amdgcn_s_wait_asynccnt(0);
#else
  asm volatile("s_wait_asynccnt 0x0" ::: "memory");
#endif
#endif
}

__device__ __forceinline__ ushort_t f2bf(float f) {
  unsigned x = __float_as_uint(f);
  unsigned r = x + 0x7FFFu + ((x >> 16) & 1u);
  return (ushort_t)(r >> 16);
}

__device__ __forceinline__ void atomic_add_f32(float* p, float v) {
  unsigned long long a = (unsigned long long)p;
  asm volatile("global_atomic_add_f32 %0, %1, off" :: "v"(a), "v"(v) : "memory");
}

// 32-byte (16 bf16) global -> LDS copy, async if available
__device__ __forceinline__ void copy32_g2l(const ushort_t* __restrict__ src,
                                           ushort_t* __restrict__ dst) {
#if defined(USE_ASYNC_LDS)
  __builtin_amdgcn_global_load_async_to_lds_b128(GPTRI(src), LPTRI(dst), 0, 0);
  __builtin_amdgcn_global_load_async_to_lds_b128(GPTRI(src), LPTRI(dst), 16, 0);
#else
  const u32x4* s = (const u32x4*)src;
  u32x4* d = (u32x4*)dst;
  d[0] = s[0]; d[1] = s[1];
#endif
}

// 16-byte global -> LDS copy, async if available
__device__ __forceinline__ void copy16_g2l(const ushort_t* __restrict__ src,
                                           ushort_t* __restrict__ dst) {
#if defined(USE_ASYNC_LDS)
  __builtin_amdgcn_global_load_async_to_lds_b128(GPTRI(src), LPTRI(dst), 0, 0);
#else
  ((u32x4*)dst)[0] = ((const u32x4*)src)[0];
#endif
}

// ---------------------------------------------------------------------------
// f32 -> bf16 conversion (grid-stride)
// ---------------------------------------------------------------------------
__global__ void cvt_f32_bf16(const float* __restrict__ src,
                             ushort_t* __restrict__ dst, int n) {
  int i = blockIdx.x * blockDim.x + threadIdx.x;
  int stride = gridDim.x * blockDim.x;
  for (; i < n; i += stride) dst[i] = f2bf(src[i]);
}

// ---------------------------------------------------------------------------
// Edge message kernel: per edge e, x = [hb[idxA[e]] | hb[idxB[e]]] (256 bf16),
// m = relu(x W1^T + b1) W2^T + b2, atomically added into msgs[idxB[e]].
// Block = 512 threads = 16 waves. Wave w owns output columns [16w, 16w+16)
// of BOTH layers and keeps its W1/W2 B-fragments resident in VGPRs.
// 32 edges per tile (two 16-row M-tiles).
// ---------------------------------------------------------------------------
__global__ __launch_bounds__(512) void edge_msg_kernel(
    const ushort_t* __restrict__ hb,    // [N][128] bf16 node states
    const ushort_t* __restrict__ Wb1,   // [256][256] bf16, row n = output col n
    const ushort_t* __restrict__ Wb2,   // [256][256] bf16
    const float*    __restrict__ b1,
    const float*    __restrict__ b2,
    const int*      __restrict__ idxA,  // first-half gather rows
    const int*      __restrict__ idxB,  // second-half gather rows + scatter target
    float*          __restrict__ msgs,  // [N][256] f32 accumulator
    int E_)
{
  __shared__ __align__(16) ushort_t xl[32 * 256];    // 16 KB x tile
  __shared__ __align__(16) ushort_t midl[32 * 256];  // 16 KB mid tile
  __shared__ int dstl[32];

  const int tid  = threadIdx.x;
  const int w    = tid >> 5;    // wave 0..15
  const int lane = tid & 31;
  const int g    = lane >> 4;   // lane-group half
  const int lm   = lane & 15;
  const int col  = (w << 4) + lm;  // output column owned by this lane

  const float b1c = b1[col];
  const float b2c = b2[col];

  // B fragment (contiguous k, same order as A): lane (n%16 + 16g) holds 16 bf16: B[32kt+16g+j][n],
  // B[k][n] = W[n][k] -> 16 contiguous bf16 of row n of W.
  Frag w1f[8], w2f[8];
#pragma unroll
  for (int kt = 0; kt < 8; ++kt) {
    const u32x4* p1 = (const u32x4*)(Wb1 + (size_t)col * 256 + kt * 32 + g * 16);
    w1f[kt].u[0] = p1[0]; w1f[kt].u[1] = p1[1];
    const u32x4* p2 = (const u32x4*)(Wb2 + (size_t)col * 256 + kt * 32 + g * 16);
    w2f[kt].u[0] = p2[0]; w2f[kt].u[1] = p2[1];
  }

  const int ntiles = (E_ + 31) >> 5;
  for (int tile = blockIdx.x; tile < ntiles; tile += gridDim.x) {
    const int ebase = tile << 5;

    // ---- gather: 512 threads each move one 32-byte chunk (async -> LDS) ----
    {
      const int e  = tid >> 4;       // edge in tile
      const int j  = tid & 15;       // 16-bf16 chunk index within 512B x row
      const int eg = ebase + e;
      const bool valid = (eg < E_);
      const int* idxp = (j < 8) ? idxA : idxB;
      const int row  = valid ? idxp[eg] : 0;
      const int xoff = (j < 8) ? (j * 16) : (128 + (j - 8) * 16);
      copy32_g2l(hb + (size_t)row * 128 + (j & 7) * 16, xl + e * 256 + xoff);
      if (j == 0) dstl[e] = valid ? idxB[eg] : 0;
    }
    wait_async_zero();
    __syncthreads();

    // ---- layer 1: acc[mt] = x_tile(mt) @ W1^T (this wave's 16 cols) --------
    v8f acc1[2];
#pragma unroll
    for (int mt = 0; mt < 2; ++mt) {
      v8f acc = {};
#pragma unroll
      for (int kt = 0; kt < 8; ++kt) {
        // A fragment: lane (m + 16g) holds A[m][32kt+16g .. +15]
        // (two adjacent 16B chunks).
        Frag a;
        const u32x4* rp = (const u32x4*)(xl + (mt * 16 + lm) * 256 + kt * 32 + g * 16);
        a.u[0] = rp[0]; a.u[1] = rp[1];
        acc = __builtin_amdgcn_wmma_f32_16x16x32_bf16(false, a.v, false, w1f[kt].v,
                                                      (short)0, acc, false, false);
      }
      acc1[mt] = acc;
    }
    // bias + ReLU -> bf16 mid tile. C/D element r of lane: M=r+8g, N=lm.
#pragma unroll
    for (int mt = 0; mt < 2; ++mt) {
#pragma unroll
      for (int r = 0; r < 8; ++r) {
        float v = acc1[mt][r] + b1c;
        v = v > 0.0f ? v : 0.0f;
        const int m = mt * 16 + r + g * 8;
        midl[m * 256 + col] = f2bf(v);
      }
    }
    __syncthreads();

    // ---- layer 2 -----------------------------------------------------------
    v8f acc2[2];
#pragma unroll
    for (int mt = 0; mt < 2; ++mt) {
      v8f acc = {};
#pragma unroll
      for (int kt = 0; kt < 8; ++kt) {
        Frag a;
        const u32x4* rp = (const u32x4*)(midl + (mt * 16 + lm) * 256 + kt * 32 + g * 16);
        a.u[0] = rp[0]; a.u[1] = rp[1];
        acc = __builtin_amdgcn_wmma_f32_16x16x32_bf16(false, a.v, false, w2f[kt].v,
                                                      (short)0, acc, false, false);
      }
      acc2[mt] = acc;
    }

    // ---- scatter-add -------------------------------------------------------
#pragma unroll
    for (int mt = 0; mt < 2; ++mt) {
#pragma unroll
      for (int r = 0; r < 8; ++r) {
        const int m = mt * 16 + r + g * 8;
        if (ebase + m < E_) {
          const float v = acc2[mt][r] + b2c;
          atomic_add_f32(msgs + (size_t)dstl[m] * 256 + col, v);
        }
      }
    }
    __syncthreads();
  }
}

// ---------------------------------------------------------------------------
// GRU kernel. 256 threads = 8 waves, 16 nodes per tile.
// gi = [msgs|feat] (384) @ W_ih^T, gh = h (128) @ W_hh^T (both 384 outputs).
// Wave w owns columns [16w,16w+16): its r/z/n gate tiles are N-tiles
// w, 8+w, 16+w, so all three gates of a column sit in the same lane slot.
// ---------------------------------------------------------------------------
__global__ __launch_bounds__(256) void gru_kernel(
    const float*    __restrict__ msgs,    // [N][256]
    const float*    __restrict__ feats,   // [N][128]
    const ushort_t* __restrict__ hb,      // [N][128] bf16
    const float*    __restrict__ hstate,  // [N][128] f32
    const ushort_t* __restrict__ Wih,     // [384][384] bf16
    const ushort_t* __restrict__ Whh,     // [384][128] bf16
    const float*    __restrict__ bih,
    const float*    __restrict__ bhh,
    float*          __restrict__ out,     // [N][128]
    int N_)
{
  __shared__ __align__(16) ushort_t gil[16 * 384];  // 12 KB gru-input tile
  __shared__ __align__(16) ushort_t hl[16 * 128];   // 4 KB h tile

  const int tid  = threadIdx.x;
  const int w    = tid >> 5;    // wave 0..7
  const int lane = tid & 31;
  const int g    = lane >> 4;
  const int lm   = lane & 15;
  const int col  = (w << 4) + lm;  // 0..127

  const float brih = bih[col], bzih = bih[128 + col], bnih = bih[256 + col];
  const float brhh = bhh[col], bzhh = bhh[128 + col], bnhh = bhh[256 + col];

  const int ntiles = (N_ + 15) >> 4;
  for (int tile = blockIdx.x; tile < ntiles; tile += gridDim.x) {
    const int nbase = tile << 4;

    // stage h tile [16][128] bf16 (raw copy, async if available)
    {
      const int nn = tid >> 4, j = tid & 15;
      const int node = nbase + nn;
      const int src_row = (node < N_) ? node : 0;
      copy16_g2l(hb + (size_t)src_row * 128 + j * 8, hl + nn * 128 + j * 8);
    }
    // stage gru_in tile [16][384] bf16 (needs f32->bf16 conversion)
    for (int idx = tid; idx < 16 * 384; idx += 256) {
      const int nn = idx / 384, c = idx % 384;
      const int node = nbase + nn;
      float v = 0.0f;
      if (node < N_)
        v = (c < 256) ? msgs[(size_t)node * 256 + c]
                      : feats[(size_t)node * 128 + (c - 256)];
      gil[idx] = f2bf(v);
    }
    wait_async_zero();
    __syncthreads();

    v8f ar = {}, az = {}, an = {}, hr = {}, hz = {}, hn = {};
#pragma unroll
    for (int kt = 0; kt < 12; ++kt) {     // K = 384 for gi
      Frag a;
      const u32x4* rp = (const u32x4*)(gil + lm * 384 + kt * 32 + g * 16);
      a.u[0] = rp[0]; a.u[1] = rp[1];
      Frag br, bz, bn;
      const u32x4* pr = (const u32x4*)(Wih + (size_t)(col)       * 384 + kt * 32 + g * 16);
      const u32x4* pz = (const u32x4*)(Wih + (size_t)(128 + col) * 384 + kt * 32 + g * 16);
      const u32x4* pn = (const u32x4*)(Wih + (size_t)(256 + col) * 384 + kt * 32 + g * 16);
      br.u[0] = pr[0]; br.u[1] = pr[1];
      bz.u[0] = pz[0]; bz.u[1] = pz[1];
      bn.u[0] = pn[0]; bn.u[1] = pn[1];
      ar = __builtin_amdgcn_wmma_f32_16x16x32_bf16(false, a.v, false, br.v, (short)0, ar, false, false);
      az = __builtin_amdgcn_wmma_f32_16x16x32_bf16(false, a.v, false, bz.v, (short)0, az, false, false);
      an = __builtin_amdgcn_wmma_f32_16x16x32_bf16(false, a.v, false, bn.v, (short)0, an, false, false);
    }
#pragma unroll
    for (int kt = 0; kt < 4; ++kt) {      // K = 128 for gh
      Frag a;
      const u32x4* rp = (const u32x4*)(hl + lm * 128 + kt * 32 + g * 16);
      a.u[0] = rp[0]; a.u[1] = rp[1];
      Frag br, bz, bn;
      const u32x4* pr = (const u32x4*)(Whh + (size_t)(col)       * 128 + kt * 32 + g * 16);
      const u32x4* pz = (const u32x4*)(Whh + (size_t)(128 + col) * 128 + kt * 32 + g * 16);
      const u32x4* pn = (const u32x4*)(Whh + (size_t)(256 + col) * 128 + kt * 32 + g * 16);
      br.u[0] = pr[0]; br.u[1] = pr[1];
      bz.u[0] = pz[0]; bz.u[1] = pz[1];
      bn.u[0] = pn[0]; bn.u[1] = pn[1];
      hr = __builtin_amdgcn_wmma_f32_16x16x32_bf16(false, a.v, false, br.v, (short)0, hr, false, false);
      hz = __builtin_amdgcn_wmma_f32_16x16x32_bf16(false, a.v, false, bz.v, (short)0, hz, false, false);
      hn = __builtin_amdgcn_wmma_f32_16x16x32_bf16(false, a.v, false, bn.v, (short)0, hn, false, false);
    }

    // gates + output. element r of lane: node = nbase + r + 8g, column = col.
#pragma unroll
    for (int r = 0; r < 8; ++r) {
      const int node = nbase + r + g * 8;
      if (node < N_) {
        const float irv = ar[r] + brih, izv = az[r] + bzih, inv = an[r] + bnih;
        const float hrv = hr[r] + brhh, hzv = hz[r] + bzhh, hnv = hn[r] + bnhh;
        const float rg = 1.0f / (1.0f + __expf(-(irv + hrv)));
        const float zg = 1.0f / (1.0f + __expf(-(izv + hzv)));
        const float ng = tanhf(inv + rg * hnv);
        const float hv = hstate[(size_t)node * 128 + col];
        out[(size_t)node * 128 + col] = (1.0f - zg) * ng + zg * hv;
      }
    }
    __syncthreads();
  }
}

// ---------------------------------------------------------------------------
// Host launcher
// ---------------------------------------------------------------------------
static inline size_t align256(size_t x) { return (x + 255) & ~(size_t)255; }

extern "C" void kernel_launch(void* const* d_in, const int* in_sizes, int n_in,
                              void* d_out, int out_size, void* d_ws, size_t ws_size,
                              hipStream_t stream) {
  (void)n_in; (void)out_size; (void)ws_size;
  const float* node_states   = (const float*)d_in[0];
  const float* node_features = (const float*)d_in[1];
  const float* W1   = (const float*)d_in[2];
  const float* b1   = (const float*)d_in[3];
  const float* W2   = (const float*)d_in[4];
  const float* b2   = (const float*)d_in[5];
  const float* rW1  = (const float*)d_in[6];
  const float* rb1  = (const float*)d_in[7];
  const float* rW2  = (const float*)d_in[8];
  const float* rb2  = (const float*)d_in[9];
  const float* W_ih = (const float*)d_in[10];
  const float* W_hh = (const float*)d_in[11];
  const float* b_ih = (const float*)d_in[12];
  const float* b_hh = (const float*)d_in[13];
  const int* from_idx = (const int*)d_in[14];
  const int* to_idx   = (const int*)d_in[15];

  const int D  = 128;
  const int N  = in_sizes[0] / D;        // 50000
  const int E  = in_sizes[14];           // 800000
  const int DM = 2 * D;                  // 256

  // workspace layout
  char* ws = (char*)d_ws;
  float*    msgs  = (float*)ws;            ws += align256((size_t)N * DM * 4);
  ushort_t* hb    = (ushort_t*)ws;         ws += align256((size_t)N * D * 2);
  ushort_t* W1b   = (ushort_t*)ws;         ws += align256((size_t)DM * DM * 2);
  ushort_t* W2b   = (ushort_t*)ws;         ws += align256((size_t)DM * DM * 2);
  ushort_t* rW1b  = (ushort_t*)ws;         ws += align256((size_t)DM * DM * 2);
  ushort_t* rW2b  = (ushort_t*)ws;         ws += align256((size_t)DM * DM * 2);
  ushort_t* Wihb  = (ushort_t*)ws;         ws += align256((size_t)(3 * D) * (3 * D) * 2);
  ushort_t* Whhb  = (ushort_t*)ws;         ws += align256((size_t)(3 * D) * D * 2);

  (void)hipMemsetAsync(msgs, 0, (size_t)N * DM * sizeof(float), stream);

  // precision conversions
  cvt_f32_bf16<<<512, 256, 0, stream>>>(node_states, hb, N * D);
  cvt_f32_bf16<<<64,  256, 0, stream>>>(W1,   W1b,  DM * DM);
  cvt_f32_bf16<<<64,  256, 0, stream>>>(W2,   W2b,  DM * DM);
  cvt_f32_bf16<<<64,  256, 0, stream>>>(rW1,  rW1b, DM * DM);
  cvt_f32_bf16<<<64,  256, 0, stream>>>(rW2,  rW2b, DM * DM);
  cvt_f32_bf16<<<128, 256, 0, stream>>>(W_ih, Wihb, (3 * D) * (3 * D));
  cvt_f32_bf16<<<64,  256, 0, stream>>>(W_hh, Whhb, (3 * D) * D);

  // forward edges: x = [h[from] | h[to]], scatter to 'to'
  edge_msg_kernel<<<2048, 512, 0, stream>>>(hb, W1b, W2b, b1, b2,
                                            from_idx, to_idx, msgs, E);
  // reverse edges: x = [h[to] | h[from]], scatter to 'from'
  edge_msg_kernel<<<2048, 512, 0, stream>>>(hb, rW1b, rW2b, rb1, rb2,
                                            to_idx, from_idx, msgs, E);

  gru_kernel<<<1024, 256, 0, stream>>>(msgs, node_features, hb, node_states,
                                       Wihb, Whhb, b_ih, b_hh,
                                       (float*)d_out, N);
}
